// GroupedSelfAttention_6485400617659
// MI455X (gfx1250) — matrix operand order repaired and checked by reading, not compiled, not run on hardware
//
#include <hip/hip_runtime.h>
#include <hip/hip_bf16.h>
#include <math.h>

#define B_   2
#define S_   2048
#define DM_  1024
#define H_   16
#define HD_  64

typedef __attribute__((ext_vector_type(16))) __bf16 v16bf;
typedef __attribute__((ext_vector_type(8)))  __bf16 v8bf;
typedef __attribute__((ext_vector_type(8)))  float  v8f;
typedef __attribute__((ext_vector_type(4)))  unsigned int v4u;
typedef __attribute__((ext_vector_type(8)))  int v8i;
typedef __attribute__((ext_vector_type(4)))  int v4i;

#if defined(__has_builtin)
#  if __has_builtin(__builtin_amdgcn_tensor_load_to_lds)
#    define HAVE_TDM 1
#  endif
#endif
#ifndef HAVE_TDM
#  define HAVE_TDM 0
#endif

static __device__ __forceinline__ v16bf cat8(v8bf lo, v8bf hi) {
  return __builtin_shufflevector(lo, hi, 0,1,2,3,4,5,6,7,8,9,10,11,12,13,14,15);
}
// A-fragment (16x32 bf16): lane = {m16, khalf}; halves 0-7 at p, halves 8-15 at p+16
static __device__ __forceinline__ v16bf lds_afrag(const __bf16* p) {
  v8bf lo = *(const v8bf*)p;
  v8bf hi = *(const v8bf*)(p + 16);
  return cat8(lo, hi);
}
// B-fragment (32x16 bf16): lane = {n16, khalf}; halves 0-7 at p, halves 8-15 at p+16
static __device__ __forceinline__ v16bf lds_bfrag(const __bf16* p) {
  v8bf lo = *(const v8bf*)p;
  v8bf hi = *(const v8bf*)(p + 16);
  return cat8(lo, hi);
}
static __device__ __forceinline__ void load16f(const float* __restrict__ p, float* t) {
  const float4* p4 = (const float4*)p;
  float4 a = p4[0], b = p4[1], c = p4[2], d = p4[3];
  t[0]=a.x; t[1]=a.y; t[2]=a.z; t[3]=a.w;
  t[4]=b.x; t[5]=b.y; t[6]=b.z; t[7]=b.w;
  t[8]=c.x; t[9]=c.y; t[10]=c.z; t[11]=c.w;
  t[12]=d.x; t[13]=d.y; t[14]=d.z; t[15]=d.w;
}
static __device__ __forceinline__ void store16bf(__bf16* dst, const float* t) {
  v8bf lo, hi;
  #pragma unroll
  for (int i=0;i<8;i++){ lo[i]=(__bf16)t[i]; hi[i]=(__bf16)t[i+8]; }
  *(v8bf*)dst       = lo;
  *(v8bf*)(dst + 8) = hi;
}

// ---------------------------------------------------------------------------
// Kernel 1: Q/K/V = x @ W^T with fused (reference-faithful) RoPE scaling.
// Writes bf16 workspace in (b, h, s, d) layout.
// Grid: (DM_/128, B_*S_/128, 3), Block: 256 (8 waves).
// ---------------------------------------------------------------------------
__global__ __launch_bounds__(256) void qkv_rope_wmma(
    const float* __restrict__ x,
    const float* __restrict__ Wq, const float* __restrict__ Wk, const float* __restrict__ Wv,
    __bf16* __restrict__ qo, __bf16* __restrict__ ko, __bf16* __restrict__ vo)
{
  __shared__ __align__(16) __bf16 sA[128][40];   // x tile: 128 rows x 32 k (pad->40)
  __shared__ __align__(16) __bf16 sB[128][40];   // W tile: 128 cols x 32 k

  const int ct = blockIdx.x, rt = blockIdx.y, which = blockIdx.z;
  const float*  __restrict__ W   = (which==0) ? Wq : ((which==1) ? Wk : Wv);
  __bf16*       __restrict__ dst = (which==0) ? qo : ((which==1) ? ko : vo);

  const int tid   = threadIdx.x;
  const int lane  = tid & 31;
  const int wave  = tid >> 5;
  const int m16   = lane & 15;
  const int khalf = lane >> 4;
  const int rw    = wave >> 1;             // wave row 0..3  (32 rows each)
  const int cw    = wave & 1;              // wave col 0..1  (64 cols each)
  const int row0  = rt * 128;
  const int col0  = ct * 128;

  const int lrow = tid >> 1;               // loader: row 0..127
  const int lk   = (tid & 1) * 16;         // loader: k seg 0/16

  v8f zero = {};
  v8f acc[2][4];
  #pragma unroll
  for (int i=0;i<2;i++) { acc[i][0]=zero; acc[i][1]=zero; acc[i][2]=zero; acc[i][3]=zero; }

  for (int kk = 0; kk < DM_; kk += 32) {
    __syncthreads();
    {
      float t[16];
      const float* srca = x + (size_t)(row0 + lrow) * DM_ + kk + lk;
      __builtin_prefetch(srca + 32, 0, 0);
      load16f(srca, t);
      store16bf(&sA[lrow][lk], t);
      const float* srcb = W + (size_t)(col0 + lrow) * DM_ + kk + lk;
      __builtin_prefetch(srcb + 32, 0, 0);
      load16f(srcb, t);
      store16bf(&sB[lrow][lk], t);
    }
    __syncthreads();

    v16bf afr[2], bfr[4];
    #pragma unroll
    for (int i=0;i<2;i++)
      afr[i] = lds_afrag(&sA[rw*32 + i*16 + m16][khalf*8]);
    #pragma unroll
    for (int j=0;j<4;j++)
      bfr[j] = lds_bfrag(&sB[cw*64 + j*16 + m16][khalf*8]);

    #pragma unroll
    for (int i=0;i<2;i++)
      #pragma unroll
      for (int j=0;j<4;j++)
        acc[i][j] = __builtin_amdgcn_wmma_f32_16x16x32_bf16(
            false, afr[i], false, bfr[j], (short)0, acc[i][j], false, false);
  }

  // Reference-faithful "rope": angle indexed by batch (uniform per block); pure scale.
  const int b = row0 >> 11;                // S_ == 2048, 128-row tiles never straddle b
  float fac[4];
  #pragma unroll
  for (int j=0;j<4;j++) {
    float fj = 1.0f;
    if (which < 2) {
      const int gcol = col0 + cw*64 + j*16 + m16;
      const int ti   = gcol & 31;
      const int top  = (gcol >> 5) & 1;
      const float theta = __powf(10000.0f, -(float)ti * (1.0f/32.0f));
      float sn, cs;
      __sincosf((float)b * theta, &sn, &cs);
      fj = top ? (cs - sn) : (cs + sn);
    }
    fac[j] = fj;
  }

  #pragma unroll
  for (int i=0;i<2;i++) {
    #pragma unroll
    for (int j=0;j<4;j++) {
      const int gcol = col0 + cw*64 + j*16 + m16;
      const int h    = gcol >> 6;
      const int dd   = gcol & (HD_-1);
      #pragma unroll
      for (int r=0;r<8;r++) {
        const int grow = row0 + rw*32 + i*16 + khalf*8 + r;
        const int s    = grow & (S_-1);
        dst[(((size_t)b * H_ + h) * S_ + s) * HD_ + dd] = (__bf16)(acc[i][j][r] * fac[j]);
      }
    }
  }
}

// ---------------------------------------------------------------------------
// Kernel 2: flash attention. Block = (b,h) x 128 q-rows; wave owns 16 q-rows.
// K tile (contiguous 8 KB bf16) fetched by the Tensor Data Mover when available;
// V staged transposed in LDS. Grid: (B_*H_, S_/128), 256 threads.
// ---------------------------------------------------------------------------
__global__ __launch_bounds__(256) void attn_wmma(
    const __bf16* __restrict__ qws, const __bf16* __restrict__ kws,
    const __bf16* __restrict__ vws, const int* __restrict__ mask,
    float* __restrict__ out)
{
  __shared__ __align__(16) __bf16 sK [64][64];       // [key][channel] (contiguous for TDM)
  __shared__ __align__(16) __bf16 sVT[64][72];       // [channel][key]
  __shared__ __align__(16) __bf16 sP [8][16][72];    // per-wave P scratch

  const int bh = blockIdx.x;
  const int qt = blockIdx.y;
  const int b  = bh >> 4;
  const int h  = bh & (H_-1);

  const int tid   = threadIdx.x;
  const int lane  = tid & 31;
  const int wave  = tid >> 5;
  const int m16   = lane & 15;
  const int khalf = lane >> 4;

  const size_t base = (size_t)bh * S_ * HD_;
  const int q0 = qt * 128 + wave * 16;

  // Q fragments (A layout) straight from bf16 global: two 16-B chunks per lane.
  v16bf qf[2];
  #pragma unroll
  for (int c=0;c<2;c++) {
    const __bf16* row = qws + base + (size_t)(q0 + m16) * HD_ + c*32 + khalf*8;
    v8bf lo = *(const v8bf*)row;
    v8bf hi = *(const v8bf*)(row + 16);
    qf[c] = cat8(lo, hi);
  }

  float mrow[8], lrow[8];
  #pragma unroll
  for (int r=0;r<8;r++){ mrow[r] = -__builtin_inff(); lrow[r] = 0.0f; }
  v8f zero = {};
  v8f o[4] = {zero, zero, zero, zero};

  const int ldr = tid >> 2;          // key row 0..63
  const int ldc = (tid & 3) * 16;    // channel seg

  for (int kb = 0; kb < S_; kb += 64) {
    __syncthreads();

#if HAVE_TDM
    if (wave == 0) {
      // Tensor DMA descriptor (ISA ch.8): one contiguous row of 4096 bf16 (8 KB).
      const unsigned long long gaddr =
          (unsigned long long)(uintptr_t)(kws + base + (size_t)kb * HD_);
      v4u g0;
      g0.x = 1u;                                         // count=1 (valid user D#)
      g0.y = (unsigned)(uintptr_t)(&sK[0][0]);           // lds_addr (bytes)
      g0.z = (unsigned)(gaddr & 0xFFFFFFFFu);            // global_addr[31:0]
      g0.w = (unsigned)((gaddr >> 32) & 0x01FFFFFFu)     // global_addr[56:32]
             | 0x80000000u;                              // type=2 ("image")
      const v8i g1 = {
        0x00010000,   // data_size=2B, no multicast/pad/iterate
        0x10000000,   // tensor_dim0[15:0]=4096 in bits[63:48]
        0x00010000,   // tensor_dim0[31:16]=0, tensor_dim1[15:0]=1
        0x10000000,   // tensor_dim1[31:16]=0, tile_dim0=4096
        0x00000001,   // tile_dim1=1, tile_dim2=0
        4096,         // tensor_dim0_stride[31:0]
        0, 0          // stride0 hi / tensor_dim1_stride
      };
      const v4i g2 = {0,0,0,0};
      const v4i g3 = {0,0,0,0};
      const v8i g4 = {0,0,0,0,0,0,0,0};   // trailing group (clang-23 6-arg form)
      __builtin_amdgcn_tensor_load_to_lds(g0, g1, g2, g3, g4, 0);
    }
#else
    {
      const __bf16* kp = kws + base + (size_t)(kb + ldr) * HD_ + ldc;
      v8bf k0 = *(const v8bf*)kp;
      v8bf k1 = *(const v8bf*)(kp + 8);
      *(v8bf*)&sK[ldr][ldc]     = k0;
      *(v8bf*)&sK[ldr][ldc + 8] = k1;
    }
#endif

    // V: bf16 global read, transposed scatter into LDS.
    {
      const __bf16* vp = vws + base + (size_t)(kb + ldr) * HD_ + ldc;
      v8bf v0 = *(const v8bf*)vp;
      v8bf v1 = *(const v8bf*)(vp + 8);
      #pragma unroll
      for (int jj=0;jj<8;jj++) {
        sVT[ldc + jj][ldr]     = v0[jj];
        sVT[ldc + 8 + jj][ldr] = v1[jj];
      }
    }

#if HAVE_TDM
    if (wave == 0) __builtin_amdgcn_s_wait_tensorcnt(0);
#endif
    __syncthreads();

    // scores: 16 q-rows x 64 keys
    v8f sfr[4];
    #pragma unroll
    for (int j=0;j<4;j++) {
      v8f sacc = zero;
      #pragma unroll
      for (int c=0;c<2;c++) {
        v16bf bk = lds_bfrag(&sK[j*16 + m16][c*32 + khalf*8]);
        sacc = __builtin_amdgcn_wmma_f32_16x16x32_bf16(
            false, qf[c], false, bk, (short)0, sacc, false, false);
      }
      sfr[j] = sacc;
    }

    // scale (1/sqrt(64)) + mask
    #pragma unroll
    for (int j=0;j<4;j++) {
      const int key   = kb + j*16 + m16;
      const bool keep = (mask[b * S_ + key] != 0);
      #pragma unroll
      for (int r=0;r<8;r++) {
        float sv = sfr[j][r] * 0.125f;
        sfr[j][r] = keep ? sv : -__builtin_inff();
      }
    }

    // online softmax (rows live across 16 lanes of each half-wave)
    #pragma unroll
    for (int r=0;r<8;r++) {
      float mx = fmaxf(fmaxf(sfr[0][r], sfr[1][r]), fmaxf(sfr[2][r], sfr[3][r]));
      mx = fmaxf(mx, __shfl_xor(mx, 1));
      mx = fmaxf(mx, __shfl_xor(mx, 2));
      mx = fmaxf(mx, __shfl_xor(mx, 4));
      mx = fmaxf(mx, __shfl_xor(mx, 8));
      const float nm = fmaxf(mrow[r], mx);
      const float alpha = (nm == -__builtin_inff()) ? 0.0f : __expf(mrow[r] - nm);
      lrow[r] *= alpha;
      #pragma unroll
      for (int jo=0;jo<4;jo++) o[jo][r] *= alpha;
      mrow[r] = nm;
    }

    // P = exp(S - m): spill to LDS (C-layout -> A-layout transpose), row sums
    float rs[8];
    #pragma unroll
    for (int r=0;r<8;r++) rs[r] = 0.0f;
    #pragma unroll
    for (int j=0;j<4;j++) {
      #pragma unroll
      for (int r=0;r<8;r++) {
        const float sv = sfr[j][r];
        const float p  = (sv > -1e30f) ? __expf(sv - mrow[r]) : 0.0f;
        rs[r] += p;
        sP[wave][khalf*8 + r][j*16 + m16] = (__bf16)p;
      }
    }
    #pragma unroll
    for (int r=0;r<8;r++) {
      float s = rs[r];
      s += __shfl_xor(s, 1);
      s += __shfl_xor(s, 2);
      s += __shfl_xor(s, 4);
      s += __shfl_xor(s, 8);
      lrow[r] += s;
    }

    // wave-private LDS write->read ordering (DS ops complete in-order per wave)
    asm volatile("s_wait_dscnt 0x0" ::: "memory");

    v16bf pf[2];
    #pragma unroll
    for (int c=0;c<2;c++)
      pf[c] = lds_afrag(&sP[wave][m16][c*32 + khalf*8]);

    // O += P x V
    #pragma unroll
    for (int jo=0;jo<4;jo++) {
      #pragma unroll
      for (int c=0;c<2;c++) {
        v16bf bv = lds_bfrag(&sVT[jo*16 + m16][c*32 + khalf*8]);
        o[jo] = __builtin_amdgcn_wmma_f32_16x16x32_bf16(
            false, pf[c], false, bv, (short)0, o[jo], false, false);
      }
    }
  }

  // epilogue: out[b, s, h, d] fp32
  #pragma unroll
  for (int jo=0;jo<4;jo++) {
    const int ch = jo*16 + m16;
    #pragma unroll
    for (int r=0;r<8;r++) {
      const int srow = q0 + khalf*8 + r;
      out[(((size_t)b * S_ + srow) * H_ + h) * HD_ + ch] = o[jo][r] / lrow[r];
    }
  }
}

extern "C" void kernel_launch(void* const* d_in, const int* in_sizes, int n_in,
                              void* d_out, int out_size, void* d_ws, size_t ws_size,
                              hipStream_t stream) {
  (void)in_sizes; (void)n_in; (void)out_size; (void)ws_size;
  const float* x    = (const float*)d_in[0];
  const int*   mask = (const int*)  d_in[1];
  const float* Wq   = (const float*)d_in[2];
  const float* Wk   = (const float*)d_in[3];
  const float* Wv   = (const float*)d_in[4];
  float* out = (float*)d_out;

  const size_t elems = (size_t)B_ * S_ * DM_;   // 4M elements per tensor
  __bf16* qws = (__bf16*)d_ws;
  __bf16* kws = qws + elems;
  __bf16* vws = kws + elems;

  dim3 g1(DM_/128, (B_*S_)/128, 3);
  qkv_rope_wmma<<<g1, 256, 0, stream>>>(x, Wq, Wk, Wv, qws, kws, vws);

  dim3 g2(B_*H_, S_/128);
  attn_wmma<<<g2, 256, 0, stream>>>(qws, kws, vws, mask, out);
}
